// MixtureLinear_3616362463382
// MI455X (gfx1250) — matrix operand order repaired and checked by reading, not compiled, not run on hardware
//
#include <hip/hip_runtime.h>
#include <hip/hip_bf16.h>

// ---------------------------------------------------------------------------
// MixtureLinear on gfx1250:
//   out[b,n,o] = sum_{c,r} x[b,n,c]*coef[n,r]*weight[o,c,r] + sum_r coef[n,r]*bias[o,r]
// Rank folded into K:  GEMM  M=B*N=8192, N=O=768, K=C*R=12288, bf16 WMMA, f32 acc.
//   A[m,k] = x[m, k/16] * coef[n(m), k%16]   (generated on the fly into LDS)
//   B[o,k] = weight viewed as row-major (O, C*R)  (K-major -> NT GEMM)
// B tiles are streamed into LDS by the Tensor Data Mover (tensor_load_to_lds),
// with descriptor padding producing the bank-conflict-free 144B row stride.
// ---------------------------------------------------------------------------

typedef __attribute__((ext_vector_type(16))) __bf16 v16bf;
typedef __attribute__((ext_vector_type(8)))  float  v8f;
typedef __attribute__((ext_vector_type(4)))  unsigned int v4u;
typedef __attribute__((ext_vector_type(8)))  int v8i;
typedef __attribute__((ext_vector_type(4)))  int v4i;

#define BATCH 8
#define NTOK  1024
#define CIN   768
#define OOUT  768
#define RANK  16
#define KDIM  (CIN * RANK)     // 12288
#define MDIM  (BATCH * NTOK)   // 8192

#define BM 128
#define BN 128
#define BK 64                  // two 16x16x32 K-steps per buffer
#define KITERS (KDIM / BK)     // 192
#define LSTR 72                // bf16 elems per LDS row (64 data + 8 pad -> 144B stride)
#define BUF_BYTES (BM * LSTR * 2)   // 18432 bytes per tile buffer

static __device__ __forceinline__ unsigned short f2b(float f) {
  __hip_bfloat16 h = __float2bfloat16(f);
  return __builtin_bit_cast(unsigned short, h);
}

// ---- kernel 1: weight f32 -> bf16 (layout preserved: (O, C*R) row-major) ----
__global__ __launch_bounds__(256) void wcvt_kernel(const float* __restrict__ w,
                                                   unsigned short* __restrict__ o) {
  size_t i = ((size_t)blockIdx.x * 256 + threadIdx.x) * 8;
  float4 a = *(const float4*)(w + i);
  float4 b = *(const float4*)(w + i + 4);
  __align__(16) unsigned short u[8];
  u[0] = f2b(a.x); u[1] = f2b(a.y); u[2] = f2b(a.z); u[3] = f2b(a.w);
  u[4] = f2b(b.x); u[5] = f2b(b.y); u[6] = f2b(b.z); u[7] = f2b(b.w);
  *(uint4*)(o + i) = *(const uint4*)u;
}

// ---- kernel 2: dterm[n,o] = sum_r coef[n,r] * bias[o,r] ----
__global__ __launch_bounds__(256) void bterm_kernel(const float* __restrict__ coef,
                                                    const float* __restrict__ bias,
                                                    float* __restrict__ d) {
  int i = blockIdx.x * 256 + threadIdx.x;      // < NTOK*OOUT
  int n = i / OOUT;
  int o = i - n * OOUT;
  const float4* cp = (const float4*)(coef + (size_t)n * RANK);
  const float4* bp = (const float4*)(bias + (size_t)o * RANK);
  float acc = 0.f;
#pragma unroll
  for (int q = 0; q < 4; ++q) {
    float4 c = cp[q], b = bp[q];
    acc += c.x * b.x + c.y * b.y + c.z * b.z + c.w * b.w;
  }
  d[i] = acc;
}

// ---- kernel 3: main bf16 WMMA GEMM with TDM-fed B tiles ----
__global__ __launch_bounds__(256) void mixture_gemm(const float* __restrict__ x,
                                                    const float* __restrict__ coef,
                                                    const unsigned short* __restrict__ wbf,
                                                    const float* __restrict__ dterm,
                                                    float* __restrict__ out) {
  // one block: A[0], A[1], B[0], B[1], each 128 rows x 72 bf16 (144B stride)
  __shared__ __align__(16) unsigned char lds_raw[4 * BUF_BYTES];
#define LDSA(buf) ((unsigned short*)(lds_raw + (buf) * BUF_BYTES))
#define LDSB(buf) ((unsigned short*)(lds_raw + 2 * BUF_BYTES + (buf) * BUF_BYTES))

  const int tid   = threadIdx.x;
  const int lane  = tid & 31;
  const int wave  = tid >> 5;        // 0..7
  const int waveM = wave & 3;        // 4 waves along M
  const int waveN = wave >> 2;       // 2 waves along N
  const int m0 = blockIdx.x * BM;    // row block in (b,n)-flattened M
  const int o0 = blockIdx.y * BN;    // col block in O
  const int n0 = m0 & (NTOK - 1);    // BM=128 divides NTOK -> block spans one batch

  // A-producer roles: 2 threads per tile row, one 32-elem (2 c-values) K-half each
  const int rowA = tid >> 1;         // 0..127
  const int half = tid & 1;          // 0: k 0..31, 1: k 32..63 of the BK slice

  // this thread's coef row lives in registers for the whole kernel
  float cf[RANK];
  {
    const float4* cp = (const float4*)(coef + (size_t)(n0 + rowA) * RANK);
    float4 c0 = cp[0], c1 = cp[1], c2 = cp[2], c3 = cp[3];
    cf[0]=c0.x; cf[1]=c0.y; cf[2]=c0.z; cf[3]=c0.w;
    cf[4]=c1.x; cf[5]=c1.y; cf[6]=c1.z; cf[7]=c1.w;
    cf[8]=c2.x; cf[9]=c2.y; cf[10]=c2.z; cf[11]=c2.w;
    cf[12]=c3.x; cf[13]=c3.y; cf[14]=c3.z; cf[15]=c3.w;
  }
  const float* xrow = x + (size_t)(m0 + rowA) * CIN;   // + (4*it + 2*half) per K-slice

  // --- TDM: DMA one 128x64 bf16 B tile (128B rows + 16B LDS pad) into LDSB(buf) ---
  auto issue_tdm = [&](int buf, int it2) {
    unsigned long long ga = (unsigned long long)(uintptr_t)wbf +
                            (((unsigned long long)o0 * KDIM) + (unsigned long long)it2 * BK) * 2ull;
    unsigned lds_off = (unsigned)(uintptr_t)(const void*)LDSB(buf);  // flat addr low 32 = LDS offset
    v4u g0;
    g0.x = 1u;                                   // count=1 (valid user descriptor)
    g0.y = lds_off;                              // lds_addr [63:32]
    g0.z = (unsigned)ga;                         // global_addr [95:64]
    g0.w = (unsigned)(ga >> 32) | (2u << 30);    // global_addr[56:32] | type=2 ("image")
    v8i g1;
    g1[0] = (1 << 16)        // data_size = 2 bytes
          | (1 << 20)        // pad_enable
          | (4 << 22)        // pad_interval: 32 DWORDs (=128B) between pads
          | (3 << 25);       // pad_amount:   4 DWORDs (=16B) of pad
    g1[1] = (int)(((unsigned)KDIM & 0xFFFFu) << 16);                       // tensor_dim0[15:0]
    g1[2] = (int)((((unsigned)KDIM >> 16) & 0xFFFFu) |
                  (((unsigned)OOUT & 0xFFFFu) << 16));                     // dim0 hi | tensor_dim1 lo
    g1[3] = (int)((((unsigned)OOUT >> 16) & 0xFFFFu) | ((unsigned)BK << 16)); // dim1 hi | tile_dim0=64
    g1[4] = (int)BN;                                                       // tile_dim1=128, tile_dim2=0
    g1[5] = (int)KDIM;                                                     // tensor_dim0_stride lo
    g1[6] = 0;                                                             // stride hi | dim1_stride lo
    g1[7] = 0;
    v4i z4 = {0, 0, 0, 0};
#if defined(__clang_major__) && (__clang_major__ >= 23)
    v8i z8 = {0, 0, 0, 0, 0, 0, 0, 0};
    __builtin_amdgcn_tensor_load_to_lds(g0, g1, z4, z4, z8, 0);
#else
    __builtin_amdgcn_tensor_load_to_lds(g0, g1, z4, z4, 0);
#endif
  };

  // --- generate one 128x64 bf16 A tile slice (this thread's 32 elements) ---
  auto produceA = [&](int buf, float2 xv) {
    __align__(16) unsigned short av[32];
#pragma unroll
    for (int r = 0; r < RANK; ++r) av[r] = f2b(xv.x * cf[r]);
#pragma unroll
    for (int r = 0; r < RANK; ++r) av[16 + r] = f2b(xv.y * cf[r]);
    uint4* pa = (uint4*)(LDSA(buf) + rowA * LSTR + half * 32);
    pa[0] = *(const uint4*)&av[0];
    pa[1] = *(const uint4*)&av[8];
    pa[2] = *(const uint4*)&av[16];
    pa[3] = *(const uint4*)&av[24];
  };

  // prologue: fill buffer 0 (TDM for B, VALU expand for A)
  if (wave == 0) issue_tdm(0, 0);
  produceA(0, *(const float2*)(xrow + half * 2));
  // distance-2 x prefetch: xnext always holds the slice for iteration (it+1)
  float2 xnext = *(const float2*)(xrow + 4 + half * 2);
  __builtin_amdgcn_s_wait_tensorcnt(0);
  __syncthreads();

  const v8f vzero = {0.f, 0.f, 0.f, 0.f, 0.f, 0.f, 0.f, 0.f};
  v8f acc[2][4];
#pragma unroll
  for (int i = 0; i < 2; ++i)
#pragma unroll
    for (int j = 0; j < 4; ++j) acc[i][j] = vzero;

  union Frag { v16bf v; uint4 q[2]; };
  const int lrow = lane & 15;
  const int aoff = (lane < 16) ? 0 : 8;    // A: lanes 0-15 hold K0..7/16..23, 16-31 hold K8..15/24..31
  const int boff = (lane < 16) ? 0 : 8;    // B: lanes 0-15 hold K0..7/16..23, 16-31 hold K8..15/24..31

  for (int it = 0; it < KITERS; ++it) {
    const int cur = it & 1;
    const bool more = (it + 1) < KITERS;

    // rotate the distance-2 x prefetch; kick next B-tile DMA before computing
    float2 xcur = xnext;
    if (it + 2 < KITERS) xnext = *(const float2*)(xrow + (it + 2) * 4 + half * 2);
    if (more && wave == 0) issue_tdm(cur ^ 1, it + 1);

    // two K-steps of fragment gathers (documented 16-bit A/B wave32 layouts) + 16 WMMAs
#pragma unroll
    for (int ks = 0; ks < 2; ++ks) {
      Frag fa[2], fb[4];
#pragma unroll
      for (int fi = 0; fi < 2; ++fi) {
        const unsigned short* p =
            LDSA(cur) + (waveM * 32 + fi * 16 + lrow) * LSTR + ks * 32;
        fa[fi].q[0] = *(const uint4*)(p + aoff);
        fa[fi].q[1] = *(const uint4*)(p + aoff + 16);
      }
#pragma unroll
      for (int fj = 0; fj < 4; ++fj) {
        const unsigned short* p =
            LDSB(cur) + (waveN * 64 + fj * 16 + lrow) * LSTR + ks * 32;
        fb[fj].q[0] = *(const uint4*)(p + boff);
        fb[fj].q[1] = *(const uint4*)(p + boff + 16);
      }
#pragma unroll
      for (int fi = 0; fi < 2; ++fi)
#pragma unroll
        for (int fj = 0; fj < 4; ++fj)
          acc[fi][fj] = __builtin_amdgcn_wmma_f32_16x16x32_bf16(
              false, fa[fi].v, false, fb[fj].v, (short)0, acc[fi][fj], false, false);
    }

    if (more) produceA(cur ^ 1, xcur);
    __builtin_amdgcn_s_wait_tensorcnt(0);  // no-op for waves that issued nothing
    __syncthreads();                       // publish B[nxt] + A[nxt] to all waves
  }

  // epilogue: D layout -> VGPR j, lanes 0-15: M=j, N=lane; lanes 16-31: M=8+j, N=lane-16
  const int rbase0 = m0 + waveM * 32 + ((lane >> 4) << 3);
  const int col0   = o0 + waveN * 64 + (lane & 15);
#pragma unroll
  for (int fi = 0; fi < 2; ++fi) {
    const int rbase = rbase0 + fi * 16;
#pragma unroll
    for (int fj = 0; fj < 4; ++fj) {
      const int col = col0 + fj * 16;
#pragma unroll
      for (int j = 0; j < 8; ++j) {
        const int row = rbase + j;               // row in flattened (b,n)
        const int n   = row & (NTOK - 1);
        out[(size_t)row * OOUT + col] = acc[fi][fj][j] + dterm[(size_t)n * OOUT + col];
      }
    }
  }
#undef LDSA
#undef LDSB
}

extern "C" void kernel_launch(void* const* d_in, const int* in_sizes, int n_in,
                              void* d_out, int out_size, void* d_ws, size_t ws_size,
                              hipStream_t stream) {
  (void)in_sizes; (void)n_in; (void)out_size; (void)ws_size;
  const float* x      = (const float*)d_in[0];
  const float* coef   = (const float*)d_in[1];
  const float* weight = (const float*)d_in[2];
  const float* bias   = (const float*)d_in[3];
  float* out = (float*)d_out;

  unsigned short* wbf = (unsigned short*)d_ws;                              // 18.9 MB bf16 weight
  float* dterm = (float*)((char*)d_ws + (size_t)OOUT * KDIM * sizeof(unsigned short)); // 3 MB bias term

  hipLaunchKernelGGL(wcvt_kernel, dim3((OOUT * KDIM) / (256 * 8)), dim3(256), 0, stream,
                     weight, wbf);
  hipLaunchKernelGGL(bterm_kernel, dim3((NTOK * OOUT) / 256), dim3(256), 0, stream,
                     coef, bias, dterm);
  hipLaunchKernelGGL(mixture_gemm, dim3(MDIM / BM, OOUT / BN), dim3(256), 0, stream,
                     x, coef, wbf, dterm, out);
}
